// ODE_RNN_72249939853933
// MI455X (gfx1250) — matrix operand order repaired and checked by reading, not compiled, not run on hardware
//
#include <hip/hip_runtime.h>
#include <hip/hip_bf16.h>
#include <math.h>

// ---------------------------------------------------------------------------
// Problem constants (from the reference): B=512, T=64, I=1024, L=512, NU=512
// ---------------------------------------------------------------------------
#define BB 512
#define TT 64
#define II 1024
#define LL 512

typedef __attribute__((ext_vector_type(8)))  _Float16 v8h;
typedef __attribute__((ext_vector_type(16))) _Float16 v16h;
typedef __attribute__((ext_vector_type(8)))  float    v8f;

// ---------------------------------------------------------------------------
// WMMA helpers
// ---------------------------------------------------------------------------
__device__ __forceinline__ v16h pack16(const _Float16* p0, const _Float16* p1) {
  v8h lo = *(const v8h*)p0;
  v8h hi = *(const v8h*)p1;
  v16h r;
#pragma unroll
  for (int i = 0; i < 8; ++i) { r[i] = lo[i]; r[8 + i] = hi[i]; }
  return r;
}

__device__ __forceinline__ v8f wmma_16x16x32(v16h a, v16h b, v8f c) {
  // D = A x B + C  (f16 inputs, f32 accumulate)
  return __builtin_amdgcn_wmma_f32_16x16x32_f16(
      /*neg_a=*/false, a, /*neg_b=*/false, b,
      /*c_mod=*/(short)0, c, /*reuse_a=*/false, /*reuse_b=*/false);
}

// ---------------------------------------------------------------------------
// GEMM: acc[M,N] = X[M,K] @ W[N,K]^T + bias   (X,W f16 row-major)
// Epilogue modes:
//   0: Yf = acc                      (f32 out)
//   1: Yh = tanh(acc)   -> f16      (gate first matmul)
//   2: Yf = sigmoid(acc)-> f32      (gate second matmul, u/r)
// Block = 128 threads = 4 waves; wave tile 32x32 (2x2 WMMA); block tile 64x64.
// Requires M,N % 64 == 0 and K % 32 == 0 (true for every call here).
// 2-stage software pipeline over K keeps loads ahead of the WMMAs.
// ---------------------------------------------------------------------------
template <int MODE>
__global__ __launch_bounds__(128) void wmma_gemm_epi_kernel(
    const _Float16* __restrict__ X, const _Float16* __restrict__ W,
    const float* __restrict__ bias, float* __restrict__ Yf,
    _Float16* __restrict__ Yh, int N, int K) {
  const int lane  = threadIdx.x & 31;
  const int wave  = threadIdx.x >> 5;
  const int mBase = blockIdx.y * 64 + (wave & 1) * 32;
  const int nBase = blockIdx.x * 64 + (wave >> 1) * 32;
  const int g     = lane >> 4;   // lane half-group
  const int l15   = lane & 15;

  v8f c00 = {}, c01 = {}, c10 = {}, c11 = {};

  // A fragment: lane (m + 16g) holds row m, K = g*8..g*8+7 and 16+g*8..
  const _Float16* a0p = X + (size_t)(mBase + l15) * K + g * 8;
  const _Float16* a1p = a0p + (size_t)16 * K;
  // B fragment: lane (n + 16g) holds W row n, K = g*8..g*8+7 and 16+g*8..
  const _Float16* b0p = W + (size_t)(nBase + l15) * K + g * 8;
  const _Float16* b1p = b0p + (size_t)16 * K;

  v16h a0 = pack16(a0p, a0p + 16);
  v16h a1 = pack16(a1p, a1p + 16);
  v16h b0 = pack16(b0p, b0p + 16);
  v16h b1 = pack16(b1p, b1p + 16);

  int kb = 0;
  for (; kb + 32 < K; kb += 32) {
    const int kn = kb + 32;
    if (kn + 32 < K) {  // stream weights ahead (global_prefetch_b8)
      __builtin_prefetch(b0p + kn + 32, 0, 1);
      __builtin_prefetch(b1p + kn + 32, 0, 1);
    }
    v16h na0 = pack16(a0p + kn, a0p + kn + 16);
    v16h na1 = pack16(a1p + kn, a1p + kn + 16);
    v16h nb0 = pack16(b0p + kn, b0p + kn + 16);
    v16h nb1 = pack16(b1p + kn, b1p + kn + 16);
    c00 = wmma_16x16x32(a0, b0, c00);
    c01 = wmma_16x16x32(a0, b1, c01);
    c10 = wmma_16x16x32(a1, b0, c10);
    c11 = wmma_16x16x32(a1, b1, c11);
    a0 = na0; a1 = na1; b0 = nb0; b1 = nb1;
  }
  c00 = wmma_16x16x32(a0, b0, c00);
  c01 = wmma_16x16x32(a0, b1, c01);
  c10 = wmma_16x16x32(a1, b0, c10);
  c11 = wmma_16x16x32(a1, b1, c11);

#pragma unroll
  for (int tm = 0; tm < 2; ++tm) {
#pragma unroll
    for (int tn = 0; tn < 2; ++tn) {
      v8f c = (tm == 0) ? (tn == 0 ? c00 : c01) : (tn == 0 ? c10 : c11);
      const int   col  = nBase + tn * 16 + l15;
      const float bv   = bias[col];
      const int   row0 = mBase + tm * 16 + g * 8;  // D: VGPR r -> row 8g + r
#pragma unroll
      for (int r = 0; r < 8; ++r) {
        const size_t idx = (size_t)(row0 + r) * N + col;
        const float  v   = c[r] + bv;
        if (MODE == 0)      Yf[idx] = v;
        else if (MODE == 1) Yh[idx] = (_Float16)tanhf(v);
        else                Yf[idx] = 1.f / (1.f + expf(-v));
      }
    }
  }
}

// ---------------------------------------------------------------------------
// Row LayerNorm + affine + leaky-relu -> f16 (ODE hidden layer, N = 1024)
// One block (256 threads) per row.
// ---------------------------------------------------------------------------
__global__ __launch_bounds__(256) void ln1_lrelu_kernel(
    const float* __restrict__ X, const float* __restrict__ gam,
    const float* __restrict__ bet, _Float16* __restrict__ outH, int N) {
  const int row = blockIdx.x;
  const float* xr = X + (size_t)row * N;
  float s = 0.f, ss = 0.f;
  for (int i = threadIdx.x; i < N; i += blockDim.x) {
    float v = xr[i]; s += v; ss += v * v;
  }
  __shared__ float rs[256], rq[256];
  rs[threadIdx.x] = s; rq[threadIdx.x] = ss;
  __syncthreads();
  for (int off = 128; off > 0; off >>= 1) {
    if (threadIdx.x < off) {
      rs[threadIdx.x] += rs[threadIdx.x + off];
      rq[threadIdx.x] += rq[threadIdx.x + off];
    }
    __syncthreads();
  }
  const float mu  = rs[0] / (float)N;
  const float var = rq[0] / (float)N - mu * mu;
  const float inv = rsqrtf(var + 1e-5f);
  for (int i = threadIdx.x; i < N; i += blockDim.x) {
    float v = (xr[i] - mu) * inv * gam[i] + bet[i];
    v = (v >= 0.f) ? v : 0.01f * v;
    outH[(size_t)row * N + i] = (_Float16)v;
  }
}

// ---------------------------------------------------------------------------
// Fused second LayerNorm + leaky-relu + RK4 combine (N = LL = 512):
//   k      = lrelu(LN(X))                       -> kout (f32, kept for later)
//   acc    = y + ca*ka + cb*kb + cc*kc + cs*k   -> outF (f32, opt) / outH (f16, opt)
// With the right coefficients this covers every RK4 stage *and* the final
// y' = y + dt*(k1 + 3(k2+k3) + k4)/8 update, so no standalone combine passes.
// ---------------------------------------------------------------------------
__global__ __launch_bounds__(256) void ln2_combine_kernel(
    const float* __restrict__ X, const float* __restrict__ gam,
    const float* __restrict__ bet, float* __restrict__ kout,
    const float* __restrict__ y, const float* __restrict__ ka,
    const float* __restrict__ kb, const float* __restrict__ kc,
    float ca, float cb, float cc, float cs,
    float* __restrict__ outF, _Float16* __restrict__ outH) {
  const int N = LL;
  const int row = blockIdx.x;
  const float* xr = X + (size_t)row * N;
  float s = 0.f, ss = 0.f;
  for (int i = threadIdx.x; i < N; i += blockDim.x) {
    float v = xr[i]; s += v; ss += v * v;
  }
  __shared__ float rs[256], rq[256];
  rs[threadIdx.x] = s; rq[threadIdx.x] = ss;
  __syncthreads();
  for (int off = 128; off > 0; off >>= 1) {
    if (threadIdx.x < off) {
      rs[threadIdx.x] += rs[threadIdx.x + off];
      rq[threadIdx.x] += rq[threadIdx.x + off];
    }
    __syncthreads();
  }
  const float mu  = rs[0] / (float)N;
  const float var = rq[0] / (float)N - mu * mu;
  const float inv = rsqrtf(var + 1e-5f);
  for (int i = threadIdx.x; i < N; i += blockDim.x) {
    const size_t idx = (size_t)row * N + i;
    float v = (xr[i] - mu) * inv * gam[i] + bet[i];
    v = (v >= 0.f) ? v : 0.01f * v;
    kout[idx] = v;
    const float acc = y[idx] + ca * ka[idx] + cb * kb[idx] + cc * kc[idx] + cs * v;
    if (outF) outF[idx] = acc;
    if (outH) outH[idx] = (_Float16)acc;
  }
}

// ---------------------------------------------------------------------------
// Elementwise kernels (grid-stride)
// ---------------------------------------------------------------------------
__global__ void cast_f32_f16_kernel(const float* __restrict__ s,
                                    _Float16* __restrict__ d, int n) {
  for (int i = blockIdx.x * blockDim.x + threadIdx.x; i < n;
       i += gridDim.x * blockDim.x)
    d[i] = (_Float16)s[i];
}

__global__ void fill0_kernel(float* __restrict__ d, int n) {
  for (int i = blockIdx.x * blockDim.x + threadIdx.x; i < n;
       i += gridDim.x * blockDim.x)
    d[i] = 0.f;
}

// xc = concat([xt, h], axis=1) -> f16 ; xt rows have stride xtStride floats
__global__ void build_xc_kernel(const float* __restrict__ xt, int xtStride,
                                const float* __restrict__ h,
                                _Float16* __restrict__ xc, int n) {
  const int W = II + LL;
  for (int i = blockIdx.x * blockDim.x + threadIdx.x; i < n;
       i += gridDim.x * blockDim.x) {
    int b = i / W, j = i - b * W;
    float v = (j < II) ? xt[(size_t)b * xtStride + j]
                       : h[(size_t)b * LL + (j - II)];
    xc[i] = (_Float16)v;
  }
}

// cc = concat([h*r, xt], axis=1) -> f16
__global__ void build_cc_kernel(const float* __restrict__ xt, int xtStride,
                                const float* __restrict__ h,
                                const float* __restrict__ r,
                                _Float16* __restrict__ cc, int n) {
  const int W = LL + II;
  for (int i = blockIdx.x * blockDim.x + threadIdx.x; i < n;
       i += gridDim.x * blockDim.x) {
    int b = i / W, j = i - b * W;
    float v = (j < LL) ? h[(size_t)b * LL + j] * r[(size_t)b * LL + j]
                       : xt[(size_t)b * xtStride + (j - LL)];
    cc[i] = (_Float16)v;
  }
}

// h = (1-u)*ns + u*h ; also emit f16 copy (input of the next ODE eval)
__global__ void gru_blend_kernel(const float* __restrict__ u,
                                 const float* __restrict__ ns,
                                 float* __restrict__ h,
                                 _Float16* __restrict__ zh, int n) {
  for (int i = blockIdx.x * blockDim.x + threadIdx.x; i < n;
       i += gridDim.x * blockDim.x) {
    float v = (1.f - u[i]) * ns[i] + u[i] * h[i];
    h[i]  = v;
    zh[i] = (_Float16)v;
  }
}

// ---------------------------------------------------------------------------
// Host orchestration
// ---------------------------------------------------------------------------
extern "C" void kernel_launch(void* const* d_in, const int* in_sizes, int n_in,
                              void* d_out, int out_size, void* d_ws,
                              size_t ws_size, hipStream_t stream) {
  (void)in_sizes; (void)n_in; (void)out_size; (void)ws_size;

  const float* x      = (const float*)d_in[0];
  const float* ug_w1  = (const float*)d_in[1];
  const float* ug_b1  = (const float*)d_in[2];
  const float* ug_w2  = (const float*)d_in[3];
  const float* ug_b2  = (const float*)d_in[4];
  const float* rg_w1  = (const float*)d_in[5];
  const float* rg_b1  = (const float*)d_in[6];
  const float* rg_w2  = (const float*)d_in[7];
  const float* rg_b2  = (const float*)d_in[8];
  const float* ns_w1  = (const float*)d_in[9];
  const float* ns_b1  = (const float*)d_in[10];
  const float* ns_w2  = (const float*)d_in[11];
  const float* ns_b2  = (const float*)d_in[12];
  const float* ode_w1 = (const float*)d_in[13];
  const float* ode_b1 = (const float*)d_in[14];
  const float* ln1_g  = (const float*)d_in[15];
  const float* ln1_b  = (const float*)d_in[16];
  const float* ode_w2 = (const float*)d_in[17];
  const float* ode_b2 = (const float*)d_in[18];
  const float* ln2_g  = (const float*)d_in[19];
  const float* ln2_b  = (const float*)d_in[20];

  // --- workspace bump allocator (256B aligned) ---
  char* ws = (char*)d_ws;
  size_t off = 0;
  auto alloc = [&](size_t bytes) -> void* {
    void* p = ws + off;
    off = (off + bytes + 255) & ~(size_t)255;
    return p;
  };

  // f16 weights
  _Float16* wug1 = (_Float16*)alloc((size_t)LL * (II + LL) * 2);
  _Float16* wrg1 = (_Float16*)alloc((size_t)LL * (II + LL) * 2);
  _Float16* wns1 = (_Float16*)alloc((size_t)LL * (II + LL) * 2);
  _Float16* wug2 = (_Float16*)alloc((size_t)LL * LL * 2);
  _Float16* wrg2 = (_Float16*)alloc((size_t)LL * LL * 2);
  _Float16* wns2 = (_Float16*)alloc((size_t)LL * LL * 2);
  _Float16* wo1  = (_Float16*)alloc((size_t)(2 * LL) * LL * 2);
  _Float16* wo2  = (_Float16*)alloc((size_t)LL * (2 * LL) * 2);
  // f16 activations
  _Float16* xc_h = (_Float16*)alloc((size_t)BB * (II + LL) * 2);
  _Float16* z_h  = (_Float16*)alloc((size_t)BB * LL * 2);
  _Float16* h1_h = (_Float16*)alloc((size_t)BB * 2 * LL * 2);
  _Float16* g1_h = (_Float16*)alloc((size_t)BB * LL * 2);
  // f32 activations
  float* h_f  = (float*)alloc((size_t)BB * LL * 4);
  float* H1_f = (float*)alloc((size_t)BB * 2 * LL * 4);
  float* H2_f = (float*)alloc((size_t)BB * LL * 4);
  float* k1   = (float*)alloc((size_t)BB * LL * 4);
  float* k2   = (float*)alloc((size_t)BB * LL * 4);
  float* k3   = (float*)alloc((size_t)BB * LL * 4);
  float* k4   = (float*)alloc((size_t)BB * LL * 4);
  float* u_f  = (float*)alloc((size_t)BB * LL * 4);
  float* r_f  = (float*)alloc((size_t)BB * LL * 4);
  float* ns_f = (float*)alloc((size_t)BB * LL * 4);

  const int nBL = BB * LL;
  auto blocks = [](int n) { int b = (n + 255) / 256; return b > 4096 ? 4096 : b; };

  auto cast = [&](const float* s, _Float16* d, int n) {
    cast_f32_f16_kernel<<<dim3(blocks(n)), dim3(256), 0, stream>>>(s, d, n);
  };
  auto gemm = [&](const _Float16* X, const _Float16* W, const float* bias,
                  float* Yf, int M, int N, int K) {
    wmma_gemm_epi_kernel<0><<<dim3(N / 64, M / 64), dim3(128), 0, stream>>>(
        X, W, bias, Yf, nullptr, N, K);
  };
  auto gemm_tanh16 = [&](const _Float16* X, const _Float16* W,
                         const float* bias, _Float16* Yh, int M, int N, int K) {
    wmma_gemm_epi_kernel<1><<<dim3(N / 64, M / 64), dim3(128), 0, stream>>>(
        X, W, bias, nullptr, Yh, N, K);
  };
  auto gemm_sig = [&](const _Float16* X, const _Float16* W, const float* bias,
                      float* Yf, int M, int N, int K) {
    wmma_gemm_epi_kernel<2><<<dim3(N / 64, M / 64), dim3(128), 0, stream>>>(
        X, W, bias, Yf, nullptr, N, K);
  };

  // --- one-time (per launch) weight conversion to f16 ---
  cast(ug_w1, wug1, LL * (II + LL));
  cast(rg_w1, wrg1, LL * (II + LL));
  cast(ns_w1, wns1, LL * (II + LL));
  cast(ug_w2, wug2, LL * LL);
  cast(rg_w2, wrg2, LL * LL);
  cast(ns_w2, wns2, LL * LL);
  cast(ode_w1, wo1, 2 * LL * LL);
  cast(ode_w2, wo2, LL * 2 * LL);

  // One ODE eval: z_h (f16) -> H2_f (pre-LN); then fused LN2+combine.
  auto ode_mm = [&]() {
    gemm(z_h, wo1, ode_b1, H1_f, BB, 2 * LL, LL);
    ln1_lrelu_kernel<<<dim3(BB), dim3(256), 0, stream>>>(H1_f, ln1_g, ln1_b,
                                                         h1_h, 2 * LL);
    gemm(h1_h, wo2, ode_b2, H2_f, BB, LL, 2 * LL);
  };
  auto ln2c = [&](float* kout, const float* ka, const float* kb,
                  const float* kc, float ca, float cb, float cc, float cs,
                  float* outF, _Float16* outH) {
    ln2_combine_kernel<<<dim3(BB), dim3(256), 0, stream>>>(
        H2_f, ln2_g, ln2_b, kout, h_f, ka, kb, kc, ca, cb, cc, cs, outF, outH);
  };

  // One RK4 step on h_f (in place). Requires z_h == (f16)h_f on entry;
  // leaves that invariant true on exit.
  auto rk4 = [&](float dt) {
    ode_mm();                                                    // k1 = f(y)
    ln2c(k1, k1, k1, k1, 0.f, 0.f, 0.f, dt / 3.f, nullptr, z_h); // y + dt*k1/3
    ode_mm();                                                    // k2
    ln2c(k2, k1, k1, k1, -dt / 3.f, 0.f, 0.f, dt, nullptr, z_h); // y+dt(k2-k1/3)
    ode_mm();                                                    // k3
    ln2c(k3, k1, k2, k2, dt, -dt, 0.f, dt, nullptr, z_h);        // y+dt(k1-k2+k3)
    ode_mm();                                                    // k4
    ln2c(k4, k1, k2, k3, dt / 8.f, 3.f * dt / 8.f, 3.f * dt / 8.f, dt / 8.f,
         h_f, z_h);                      // y' = y + dt*(k1+3(k2+k3)+k4)/8
  };

  // GRU update: h_f = gru(h_f, xt); also refreshes z_h = (f16)h_f.
  auto gru = [&](const float* xt_base) {
    const int nXC = BB * (II + LL);
    build_xc_kernel<<<dim3(blocks(nXC)), dim3(256), 0, stream>>>(
        xt_base, TT * II, h_f, xc_h, nXC);
    gemm_tanh16(xc_h, wug1, ug_b1, g1_h, BB, LL, II + LL);  // tanh fused
    gemm_sig(g1_h, wug2, ug_b2, u_f, BB, LL, LL);           // sigmoid fused
    gemm_tanh16(xc_h, wrg1, rg_b1, g1_h, BB, LL, II + LL);
    gemm_sig(g1_h, wrg2, rg_b2, r_f, BB, LL, LL);
    build_cc_kernel<<<dim3(blocks(nXC)), dim3(256), 0, stream>>>(
        xt_base, TT * II, h_f, r_f, xc_h, nXC);
    gemm_tanh16(xc_h, wns1, ns_b1, g1_h, BB, LL, II + LL);
    gemm(g1_h, wns2, ns_b2, ns_f, BB, LL, LL);
    gru_blend_kernel<<<dim3(blocks(nBL)), dim3(256), 0, stream>>>(
        u_f, ns_f, h_f, z_h, nBL);
  };

  // --- full sequence (x is scanned in reverse time order) ---
  fill0_kernel<<<dim3(blocks(nBL)), dim3(256), 0, stream>>>(h_f, nBL);
  gru(x + (size_t)(TT - 1) * II);  // z0 = gru(0, x_rev[:,0,:])
  for (int s = 0; s < TT; ++s) {
    rk4(0.5f);
    rk4(0.5f);
    gru(x + (size_t)(TT - 1 - s) * II);  // xt = x_rev[:, s, :]
  }

  hipMemcpyAsync(d_out, h_f, (size_t)BB * LL * sizeof(float),
                 hipMemcpyDeviceToDevice, stream);
}
